// BLSTM_02_36936718745608
// MI455X (gfx1250) — compile-verified, matrix operand order checked by reading, not run on hardware
//
#include <hip/hip_runtime.h>

// ---------------------------------------------------------------------------
// Types for CDNA5 WMMA (wave32): bf16 A/B operands (16 bf16 = 8 VGPRs), f32 acc
// ---------------------------------------------------------------------------
typedef __bf16 v16bf __attribute__((ext_vector_type(16)));
typedef float  v8f   __attribute__((ext_vector_type(8)));

union ABReg { v16bf v; unsigned u[8]; };

#define H_       700
#define HP_      704      // padded hidden
#define GP_      2816     // 4*HP_ padded gate width
#define B_       64
#define T_       512
#define M_       32768    // B_*T_
#define IN_      201
#define INP_     224      // padded input (multiple of 32)
#define CATP_    1408     // 2*HP_ padded concat
#define NCOLWG_  44       // HP_/16 column-tile workgroups per direction
#define KCH_     22       // HP_/32 K-chunks for recurrent GEMM
#define LDSW_    (KCH_*4*256)  // dwords of swizzled w_hh per WG

__device__ __forceinline__ unsigned short f2bf(float f) {
  unsigned u = __float_as_uint(f);
  u += 0x7fffu + ((u >> 16) & 1u);      // round-to-nearest-even
  return (unsigned short)(u >> 16);
}
__device__ __forceinline__ float bf2f(unsigned short h) {
  return __uint_as_float(((unsigned)h) << 16);
}

// K index of the even element held in VGPR v for lane-half g (ISA 7.12.2,
// 16-bit A 16x32 layout; B mirrors with N<->M roles). Pair covers K, K+1.
__device__ __forceinline__ int kidx(int v, int g) {
  return (v < 4 ? 2 * v : 16 + 2 * (v - 4)) + 8 * g;
}

// ---------------------------------------------------------------------------
// Pack kernels: fp32 weights -> bf16, pre-swizzled into WMMA B-operand tiles.
// Swizzled layout: tile (kc, nt) -> 256 dwords, dword = [lane*8 + v],
// value = {B[kc*32+kidx(v,g)][n] , B[..+1][n]} with n = nt*16 + lane%16.
// ---------------------------------------------------------------------------
__device__ __forceinline__ unsigned fetch_bf(const float* __restrict__ src,
                                             int mode, int k, int n,
                                             int segLen, int segPad, int Isrc) {
  float val = 0.f;
  if (mode == 0) {              // w_ih^T with segment remap (concat padding)
    int grp = n / HP_, j = n % HP_;
    int seg = k / segPad, jk = k % segPad;
    if (j < H_ && jk < segLen)
      val = src[(size_t)(grp * H_ + j) * Isrc + seg * segLen + jk];
  } else if (mode == 1) {       // w_hh^T
    int grp = n / HP_, j = n % HP_;
    if (k < H_ && j < H_)
      val = src[(size_t)(grp * H_ + j) * H_ + k];
  } else {                      // lin_w^T : [IN_, H_] -> [HP_][256]
    if (k < H_ && n < IN_)
      val = src[(size_t)n * H_ + k];
  }
  return (unsigned)f2bf(val);
}

__global__ void pack_b_sw(const float* __restrict__ src, unsigned* __restrict__ dst,
                          int mode, int Kpad, int Npad,
                          int segLen, int segPad, int Isrc) {
  int total = (Kpad >> 5) * (Npad >> 4) * 256;
  int idx = blockIdx.x * 256 + threadIdx.x;
  if (idx >= total) return;
  int tIdx = idx >> 8, off = idx & 255;
  int lane = off >> 3, v = off & 7;
  int ntiles = Npad >> 4;
  int kc = tIdx / ntiles, nt = tIdx % ntiles;
  int g = lane >> 4;
  int col = nt * 16 + (lane & 15);
  int k = kc * 32 + kidx(v, g);
  unsigned lo = fetch_bf(src, mode, k,     col, segLen, segPad, Isrc);
  unsigned hi = fetch_bf(src, mode, k + 1, col, segLen, segPad, Isrc);
  dst[idx] = lo | (hi << 16);
}

__global__ void pack_x_bf16(const float* __restrict__ x, unsigned short* __restrict__ dst) {
  int idx = blockIdx.x * 256 + threadIdx.x;
  if (idx >= M_ * INP_) return;
  int m = idx / INP_, k = idx % INP_;
  dst[idx] = (k < IN_) ? f2bf(x[(size_t)m * IN_ + k]) : (unsigned short)0;
}

__global__ void pack_bias(const float* __restrict__ b_ih, const float* __restrict__ b_hh,
                          float* __restrict__ dst) {
  int n = blockIdx.x * 256 + threadIdx.x;
  if (n >= GP_) return;
  int grp = n / HP_, j = n % HP_;
  dst[n] = (j < H_) ? (b_ih[grp * H_ + j] + b_hh[grp * H_ + j]) : 0.f;
}

__global__ void pack_linb(const float* __restrict__ b, float* __restrict__ dst) {
  int n = blockIdx.x * 256 + threadIdx.x;
  if (n >= 256) return;
  dst[n] = (n < IN_) ? b[n] : 0.f;
}

__global__ void sum_dirs(const unsigned short* __restrict__ act,
                         unsigned short* __restrict__ s) {
  int idx = blockIdx.x * 256 + threadIdx.x;
  if (idx >= M_ * HP_) return;
  int m = idx / HP_, j = idx % HP_;
  s[idx] = f2bf(bf2f(act[(size_t)m * CATP_ + j]) +
                bf2f(act[(size_t)m * CATP_ + HP_ + j]));
}

// ---------------------------------------------------------------------------
// Generic bf16 WMMA GEMM: C[M x Nreal] = A[M x K](bf16,row-major) * Bsw + bias
// Block = 128 threads (4 waves); each wave: 16 rows x 64 cols (4 WMMA tiles).
// ---------------------------------------------------------------------------
__global__ void __launch_bounds__(128)
gemm_bf16_wmma(const unsigned short* __restrict__ A, int lda,
               const unsigned* __restrict__ Bsw, int ntilesN,
               const float* __restrict__ bias,
               void* __restrict__ Cout, int ldc, int Nreal, int storeBf16, int K) {
  const int wv = threadIdx.x >> 5;
  const int lane = threadIdx.x & 31;
  const int lrow = lane & 15, g = lane >> 4;
  const int m0 = blockIdx.y * 64 + wv * 16;
  const size_t arow = (size_t)(m0 + lrow);

  v8f acc[4];
#pragma unroll
  for (int nt = 0; nt < 4; ++nt) acc[nt] = (v8f){0,0,0,0,0,0,0,0};

  const int nKc = K >> 5;
  for (int kc = 0; kc < nKc; ++kc) {
    ABReg a;
#pragma unroll
    for (int v = 0; v < 8; ++v)
      a.u[v] = *(const unsigned*)(A + arow * lda + (kc * 32 + kidx(v, g)));
#pragma unroll
    for (int nt = 0; nt < 4; ++nt) {
      ABReg b;
      const unsigned* bp = Bsw + ((size_t)kc * ntilesN + (blockIdx.x * 4 + nt)) * 256 + lane * 8;
#pragma unroll
      for (int v = 0; v < 8; ++v) b.u[v] = bp[v];
      acc[nt] = __builtin_amdgcn_wmma_f32_16x16x32_bf16(
          false, a.v, false, b.v, (short)0, acc[nt], false, false);
    }
  }
#pragma unroll
  for (int nt = 0; nt < 4; ++nt) {
    int n = blockIdx.x * 64 + nt * 16 + lrow;
    if (n >= Nreal) continue;
    float bs = bias[n];
#pragma unroll
    for (int r = 0; r < 8; ++r) {
      size_t m = (size_t)(m0 + r + 8 * g);
      float val = acc[nt][r] + bs;
      if (storeBf16) ((unsigned short*)Cout)[m * ldc + n] = f2bf(val);
      else           ((float*)Cout)[m * ldc + n] = val;
    }
  }
}

// ---------------------------------------------------------------------------
// Grid barrier (monotonic counter, per direction)
// ---------------------------------------------------------------------------
__device__ __forceinline__ void grid_barrier(unsigned* cnt, unsigned expected) {
  __threadfence();
  __syncthreads();
  if (threadIdx.x == 0) {
    atomicAdd(cnt, 1u);
    while (atomicAdd(cnt, 0u) < expected) __builtin_amdgcn_s_sleep(2);
  }
  __syncthreads();
}

// ---------------------------------------------------------------------------
// Persistent recurrent kernel: grid (44, 2 dirs), block 128 (4 waves).
// WG bx owns j-columns [16*bx, 16*bx+16) of all 4 gate groups.
// w_hh^T tiles for those columns live in LDS for all 512 steps.
// h double-buffered in global; one grid barrier per step.
// ---------------------------------------------------------------------------
__global__ void __launch_bounds__(128)
lstm_recurrent(const unsigned short* __restrict__ xgF,
               const unsigned short* __restrict__ xgB,
               const unsigned* __restrict__ whhF,
               const unsigned* __restrict__ whhB,
               unsigned short* __restrict__ hbuf,    // [2 dir][2 phase][64*HP_]
               unsigned short* __restrict__ actOut,  // [M_][CATP_]
               unsigned* __restrict__ barCnt) {      // [2]
  extern __shared__ unsigned Ls[];                   // LDSW_ dwords (88 KiB)
  const int dir = blockIdx.y;
  const int bx = blockIdx.x;
  const unsigned short* xg = dir ? xgB : xgF;
  const unsigned* wsw = dir ? whhB : whhF;
  unsigned short* hB = hbuf + (size_t)dir * (2 * B_ * HP_);
  unsigned* cnt = barCnt + dir;

  // Stage this WG's w_hh^T tiles (all 22 K-chunks x 4 gate groups) into LDS.
  for (int i = threadIdx.x; i < LDSW_; i += 128) {
    int kc = i >> 10, grp = (i >> 8) & 3, off = i & 255;
    Ls[i] = wsw[((size_t)kc * (GP_ / 16) + grp * NCOLWG_ + bx) * 256 + off];
  }
  // Zero both h phases (done by bx==0 of each direction).
  if (bx == 0) {
    unsigned* hz = (unsigned*)hB;
    for (int i = threadIdx.x; i < (2 * B_ * HP_) / 2; i += 128) hz[i] = 0u;
  }
  unsigned expect = NCOLWG_;
  grid_barrier(cnt, expect);

  const int wv = threadIdx.x >> 5;
  const int lane = threadIdx.x & 31;
  const int lrow = lane & 15, g = lane >> 4;
  const int arow = wv * 16 + lrow;       // batch row for A loads
  const int jcol = bx * 16 + lrow;       // hidden column for C/D + state

  float cst[8];
#pragma unroll
  for (int r = 0; r < 8; ++r) cst[r] = 0.f;

  for (int s = 0; s < T_; ++s) {
    const int t = dir ? (T_ - 1 - s) : s;
    const unsigned short* hr = hB + (size_t)(s & 1) * (B_ * HP_);
    unsigned short* hw = hB + (size_t)((s + 1) & 1) * (B_ * HP_);

    v8f acc[4];
#pragma unroll
    for (int q = 0; q < 4; ++q) acc[q] = (v8f){0,0,0,0,0,0,0,0};

    for (int kc = 0; kc < KCH_; ++kc) {
      ABReg a;
#pragma unroll
      for (int v = 0; v < 8; ++v)
        a.u[v] = *(const unsigned*)(hr + (size_t)arow * HP_ + kc * 32 + kidx(v, g));
#pragma unroll
      for (int grp = 0; grp < 4; ++grp) {
        ABReg b;
        const unsigned* bp = Ls + (kc * 4 + grp) * 256 + lane * 8;
#pragma unroll
        for (int v = 0; v < 8; ++v) b.u[v] = bp[v];
        acc[grp] = __builtin_amdgcn_wmma_f32_16x16x32_bf16(
            false, a.v, false, b.v, (short)0, acc[grp], false, false);
      }
    }

#pragma unroll
    for (int r = 0; r < 8; ++r) {
      const int brow = wv * 16 + r + 8 * g;     // C/D layout row
      const size_t xbase = ((size_t)brow * T_ + t) * GP_;
      float gi = acc[0][r] + bf2f(xg[xbase + jcol]);
      float gf = acc[1][r] + bf2f(xg[xbase + HP_ + jcol]);
      float gc = acc[2][r] + bf2f(xg[xbase + 2 * HP_ + jcol]);
      float go = acc[3][r] + bf2f(xg[xbase + 3 * HP_ + jcol]);
      float iv = 1.f / (1.f + __expf(-gi));
      float fv = 1.f / (1.f + __expf(-gf));
      float gv = tanhf(gc);
      float ov = 1.f / (1.f + __expf(-go));
      float c = fv * cst[r] + iv * gv;
      cst[r] = c;
      unsigned short hb16 = f2bf(ov * tanhf(c));
      hw[(size_t)brow * HP_ + jcol] = hb16;
      actOut[((size_t)brow * T_ + t) * CATP_ + (size_t)dir * HP_ + jcol] = hb16;
    }
    expect += NCOLWG_;
    grid_barrier(cnt, expect);
  }
}

// ---------------------------------------------------------------------------
// Host orchestration
// ---------------------------------------------------------------------------
extern "C" void kernel_launch(void* const* d_in, const int* in_sizes, int n_in,
                              void* d_out, int out_size, void* d_ws, size_t ws_size,
                              hipStream_t stream) {
  const float* x = (const float*)d_in[0];
  const float *wih[3][2], *whh[3][2], *bih[3][2], *bhh[3][2];
  for (int l = 0; l < 3; ++l)
    for (int d = 0; d < 2; ++d) {
      int base = 1 + (l * 2 + d) * 4;
      wih[l][d] = (const float*)d_in[base + 0];
      whh[l][d] = (const float*)d_in[base + 1];
      bih[l][d] = (const float*)d_in[base + 2];
      bhh[l][d] = (const float*)d_in[base + 3];
    }
  const float* lin_w = (const float*)d_in[25];
  const float* lin_b = (const float*)d_in[26];

  char* p = (char*)d_ws;
  auto alloc = [&](size_t bytes) -> void* {
    void* r = (void*)p;
    p += (bytes + 255) & ~(size_t)255;
    return r;
  };
  unsigned* cnt          = (unsigned*)alloc(256);
  unsigned short* x_bf   = (unsigned short*)alloc((size_t)M_ * INP_ * 2);
  unsigned short* actA   = (unsigned short*)alloc((size_t)M_ * CATP_ * 2);
  unsigned short* actB   = (unsigned short*)alloc((size_t)M_ * CATP_ * 2);
  unsigned short* xg0    = (unsigned short*)alloc((size_t)M_ * GP_ * 2);
  unsigned short* xg1    = (unsigned short*)alloc((size_t)M_ * GP_ * 2);
  unsigned short* sums   = (unsigned short*)alloc((size_t)M_ * HP_ * 2);
  unsigned short* hbuf   = (unsigned short*)alloc((size_t)2 * 2 * B_ * HP_ * 2);
  unsigned* wihsw[3][2];
  unsigned* whhsw[3][2];
  float* biasg[3][2];
  for (int l = 0; l < 3; ++l)
    for (int d = 0; d < 2; ++d) {
      int kc = (l == 0 ? INP_ : CATP_) >> 5;
      wihsw[l][d] = (unsigned*)alloc((size_t)kc * (GP_ / 16) * 256 * 4);
      whhsw[l][d] = (unsigned*)alloc((size_t)KCH_ * (GP_ / 16) * 256 * 4);
      biasg[l][d] = (float*)alloc((size_t)GP_ * 4);
    }
  unsigned* linsw = (unsigned*)alloc((size_t)KCH_ * 16 * 256 * 4);
  float* linbp    = (float*)alloc(256 * 4);

  // ---- weight / input packing (all bf16, B pre-swizzled to WMMA layout) ----
  {
    int n = M_ * INP_;
    pack_x_bf16<<<(n + 255) / 256, 256, 0, stream>>>(x, x_bf);
  }
  for (int l = 0; l < 3; ++l)
    for (int d = 0; d < 2; ++d) {
      int Kpad = (l == 0) ? INP_ : CATP_;
      int segLen = (l == 0) ? IN_ : H_;
      int segPad = (l == 0) ? INP_ : HP_;
      int Isrc = (l == 0) ? IN_ : 2 * H_;
      int tot = (Kpad >> 5) * (GP_ >> 4) * 256;
      pack_b_sw<<<(tot + 255) / 256, 256, 0, stream>>>(
          wih[l][d], wihsw[l][d], 0, Kpad, GP_, segLen, segPad, Isrc);
      tot = KCH_ * (GP_ >> 4) * 256;
      pack_b_sw<<<(tot + 255) / 256, 256, 0, stream>>>(
          whh[l][d], whhsw[l][d], 1, HP_, GP_, 1, 1, 1);
      pack_bias<<<(GP_ + 255) / 256, 256, 0, stream>>>(bih[l][d], bhh[l][d], biasg[l][d]);
    }
  {
    int tot = KCH_ * 16 * 256;
    pack_b_sw<<<(tot + 255) / 256, 256, 0, stream>>>(lin_w, linsw, 2, HP_, 256, 1, 1, 1);
    pack_linb<<<1, 256, 0, stream>>>(lin_b, linbp);
  }

  // ---- layers ----
  const unsigned short* inA = x_bf;
  int lda = INP_, K = INP_;
  unsigned short* acts[2] = {actA, actB};
  for (int l = 0; l < 3; ++l) {
    dim3 gg(GP_ / 64, M_ / 64);  // (44, 512)
    gemm_bf16_wmma<<<gg, 128, 0, stream>>>(inA, lda, wihsw[l][0], GP_ / 16,
                                           biasg[l][0], xg0, GP_, GP_, 1, K);
    gemm_bf16_wmma<<<gg, 128, 0, stream>>>(inA, lda, wihsw[l][1], GP_ / 16,
                                           biasg[l][1], xg1, GP_, GP_, 1, K);
    hipMemsetAsync(cnt, 0, 256, stream);  // reset per-direction barrier counters
    unsigned short* outAct = acts[l & 1];
    lstm_recurrent<<<dim3(NCOLWG_, 2), 128, LDSW_ * 4, stream>>>(
        xg0, xg1, whhsw[l][0], whhsw[l][1], hbuf, outAct, cnt);
    inA = outAct;
    lda = CATP_;
    K = CATP_;
  }

  // ---- hf + hb, then final linear ----
  {
    int n = M_ * HP_;
    sum_dirs<<<(n + 255) / 256, 256, 0, stream>>>(inA, sums);
  }
  {
    dim3 gg(256 / 64, M_ / 64);
    gemm_bf16_wmma<<<gg, 128, 0, stream>>>(sums, HP_, linsw, 16, linbp,
                                           d_out, IN_, IN_, 0, HP_);
  }
}
